// Qwen2MoeSparseMoeBlock_51333449121906
// MI455X (gfx1250) — compile-verified, matrix operand order checked by reading, not run on hardware
//
#include <hip/hip_runtime.h>
#include <cstdint>

// ---------------------------------------------------------------------------
// Qwen2-MoE sparse block for MI455X (gfx1250).
// bf16 WMMA GEMMs, double-buffered LDS, async global->LDS tile staging.
// ---------------------------------------------------------------------------

typedef __bf16 bf16;
typedef __attribute__((ext_vector_type(16))) __bf16 v16bf;
typedef __attribute__((ext_vector_type(8)))  __bf16 v8bf;
typedef __attribute__((ext_vector_type(8)))  float  v8f;
typedef int v4i __attribute__((vector_size(16)));   // matches builtin's int4

#define T_TOK 1024
#define HDIM  2048
#define NEXP  8
#define TOPK  4
#define IE_SZ 1408
#define IS_SZ 5632
#define KK_TOT (IS_SZ + NEXP * IE_SZ)   // 16896 (div by 32; IS_SZ too)

#define TM 128
#define TN 64
#define TK 32
#define LDS_LD 48   // bf16 elems/row: 96B stride (16B aligned)

// ---- gfx1250 async global->LDS (ASYNCcnt) with safe fallback --------------
#if defined(__gfx1250__) && __has_builtin(__builtin_amdgcn_global_load_async_to_lds_b128)
#define HAS_ASYNC_LDS 1
#else
#define HAS_ASYNC_LDS 0
#endif

#if HAS_ASYNC_LDS
#define ASYNC_CP16(gsrc, ldst)                                                   \
    __builtin_amdgcn_global_load_async_to_lds_b128(                              \
        (__attribute__((address_space(1))) v4i*)(uintptr_t)(gsrc),               \
        (__attribute__((address_space(3))) v4i*)(uintptr_t)(ldst), 0, 0)
#else
#define ASYNC_CP16(gsrc, ldst) (*(uint4*)(ldst) = *(const uint4*)(gsrc))
#endif

#if HAS_ASYNC_LDS
#if __has_builtin(__builtin_amdgcn_s_wait_asynccnt)
#define WAIT_ASYNC() __builtin_amdgcn_s_wait_asynccnt(0)
#else
#define WAIT_ASYNC() asm volatile("s_wait_asynccnt 0x0" ::: "memory")
#endif
#else
#define WAIT_ASYNC() ((void)0)
#endif

union AFrag { v16bf v; v8bf h[2]; };
union PkBf  { bf16 h[2]; unsigned int u; };

// ---------------------------------------------------------------------------
// Kernel 1: router (softmax -> top-4) + shared sigmoid gate. One wave/token.
// ---------------------------------------------------------------------------
__global__ __launch_bounds__(128)
void router_kernel(const float* __restrict__ x,
                   const float* __restrict__ w_gate,   // [H, 8]
                   const float* __restrict__ w_sgate,  // [H, 1]
                   float* __restrict__ comb,           // [T, 8]
                   float* __restrict__ gate_sig)       // [T]
{
    const int warp = threadIdx.x >> 5;
    const int lane = threadIdx.x & 31;
    const int t = blockIdx.x * 4 + warp;
    if (t >= T_TOK) return;

    float acc[NEXP];
#pragma unroll
    for (int e = 0; e < NEXP; ++e) acc[e] = 0.f;
    float accg = 0.f;

    const float* xr = x + (long)t * HDIM;
    for (int h = lane; h < HDIM; h += 32) {
        const float xv = xr[h];
        const float* wr = w_gate + (long)h * NEXP;
#pragma unroll
        for (int e = 0; e < NEXP; ++e) acc[e] += xv * wr[e];
        accg += xv * w_sgate[h];
    }
#pragma unroll
    for (int off = 16; off > 0; off >>= 1) {
#pragma unroll
        for (int e = 0; e < NEXP; ++e) acc[e] += __shfl_down(acc[e], off, 32);
        accg += __shfl_down(accg, off, 32);
    }

    if (lane == 0) {
        float m = acc[0];
#pragma unroll
        for (int e = 1; e < NEXP; ++e) m = fmaxf(m, acc[e]);
        float p[NEXP], s = 0.f;
#pragma unroll
        for (int e = 0; e < NEXP; ++e) { p[e] = __expf(acc[e] - m); s += p[e]; }
        const float inv = 1.f / s;
#pragma unroll
        for (int e = 0; e < NEXP; ++e) p[e] *= inv;

        float c[NEXP];
#pragma unroll
        for (int e = 0; e < NEXP; ++e) c[e] = 0.f;
#pragma unroll
        for (int k = 0; k < TOPK; ++k) {
            int bj = 0; float bv = p[0];
#pragma unroll
            for (int e = 1; e < NEXP; ++e)
                if (p[e] > bv) { bv = p[e]; bj = e; }
            c[bj] = bv; p[bj] = -1.f;
        }
#pragma unroll
        for (int e = 0; e < NEXP; ++e) comb[(long)t * NEXP + e] = c[e];
        gate_sig[t] = 1.f / (1.f + __expf(-accg));
    }
}

// ---------------------------------------------------------------------------
// Kernel 2: fp32 -> bf16 conversion of the token activations.
// ---------------------------------------------------------------------------
__global__ __launch_bounds__(256)
void cvt_bf16_kernel(const float* __restrict__ x, bf16* __restrict__ xb, long n)
{
    const long i = ((long)blockIdx.x * blockDim.x + threadIdx.x) * 4;
    if (i + 3 < n) {
        const float4 v = *(const float4*)(x + i);
        xb[i + 0] = (bf16)v.x;  xb[i + 1] = (bf16)v.y;
        xb[i + 2] = (bf16)v.z;  xb[i + 3] = (bf16)v.w;
    }
}

// ---------------------------------------------------------------------------
// Kernel 3: fused SwiGLU GEMM:  act = scale(t) * silu(X@Wg) * (X@Wu)
// Shared expert (grid.z=1) and routed experts (grid.z=8) share this kernel.
// Double-buffered, X tile via async-to-LDS, W tiles register-staged fp32->bf16.
// ---------------------------------------------------------------------------
__global__ __launch_bounds__(256)
void gu_act_kernel(const bf16*  __restrict__ Xb,      // [T, H] bf16
                   const float* __restrict__ W,
                   long  w_estride, int Ihalf,
                   const float* __restrict__ scale, int scale_stride,
                   bf16* __restrict__ Aout, int out_col0, int out_estride)
{
    __shared__ bf16 Xs[2][TM * LDS_LD];
    __shared__ bf16 Gs[2][TN * LDS_LD];
    __shared__ bf16 Us[2][TN * LDS_LD];

    const int eidx = blockIdx.z;
    const float* Wp = W + (long)eidx * w_estride;
    const float* sc = scale + eidx;
    const int ldw = 2 * Ihalf;
    const int n0 = blockIdx.x * TN;            // in "half" column space
    const int m0 = blockIdx.y * TM;
    const int outc0 = out_col0 + eidx * out_estride;

    const int tid  = threadIdx.x;
    const int lane = tid & 31;
    const int wv   = tid >> 5;
    const int mw   = wv & 3;                   // 0..3 : M 32-block
    const int nw   = wv >> 2;                  // 0..1 : N 32-block

    // X copy geometry: two 16B chunks/thread -> rows xr and xr+64, k-chunk xk.
    const int xr = tid >> 2;                   // 0..63
    const int xk = (tid & 3) * 8;              // 0,8,16,24
    const bf16* xg0 = Xb + (long)(m0 + xr) * HDIM + xk;
    const bf16* xg1 = Xb + (long)(m0 + xr + 64) * HDIM + xk;

    // W copy geometry: adjacent rows wk2,wk2+1; 4 consecutive n at wnn.
    const int wk2 = (tid >> 4) * 2;            // 0,2,..,30
    const int wnn = (tid & 15) * 4;            // 0..60

    auto issue_x = [&](int s, int kb) {
        ASYNC_CP16(xg0 + kb, &Xs[s][xr * LDS_LD + xk]);
        ASYNC_CP16(xg1 + kb, &Xs[s][(xr + 64) * LDS_LD + xk]);
    };

    struct WStage { float4 g0, g1, u0, u1; };
    auto load_w = [&](int kb) {
        WStage v;
        const float* r0 = Wp + (long)(kb + wk2) * ldw + n0 + wnn;
        v.g0 = *(const float4*)(r0);
        v.u0 = *(const float4*)(r0 + Ihalf);
        v.g1 = *(const float4*)(r0 + ldw);
        v.u1 = *(const float4*)(r0 + ldw + Ihalf);
        return v;
    };
    auto store_w = [&](int s, const WStage& v) {
        // transposed [n][k], two adjacent k packed per 4B store
        bf16* gp = &Gs[s][wnn * LDS_LD + wk2];
        bf16* up = &Us[s][wnn * LDS_LD + wk2];
        PkBf t0, t1;
        t0.h[0] = (bf16)v.g0.x; t0.h[1] = (bf16)v.g1.x; *(unsigned int*)(gp + 0 * LDS_LD) = t0.u;
        t0.h[0] = (bf16)v.g0.y; t0.h[1] = (bf16)v.g1.y; *(unsigned int*)(gp + 1 * LDS_LD) = t0.u;
        t0.h[0] = (bf16)v.g0.z; t0.h[1] = (bf16)v.g1.z; *(unsigned int*)(gp + 2 * LDS_LD) = t0.u;
        t0.h[0] = (bf16)v.g0.w; t0.h[1] = (bf16)v.g1.w; *(unsigned int*)(gp + 3 * LDS_LD) = t0.u;
        t1.h[0] = (bf16)v.u0.x; t1.h[1] = (bf16)v.u1.x; *(unsigned int*)(up + 0 * LDS_LD) = t1.u;
        t1.h[0] = (bf16)v.u0.y; t1.h[1] = (bf16)v.u1.y; *(unsigned int*)(up + 1 * LDS_LD) = t1.u;
        t1.h[0] = (bf16)v.u0.z; t1.h[1] = (bf16)v.u1.z; *(unsigned int*)(up + 2 * LDS_LD) = t1.u;
        t1.h[0] = (bf16)v.u0.w; t1.h[1] = (bf16)v.u1.w; *(unsigned int*)(up + 3 * LDS_LD) = t1.u;
    };

    v8f accG[2][2], accU[2][2];
#pragma unroll
    for (int mi = 0; mi < 2; ++mi)
#pragma unroll
        for (int ni = 0; ni < 2; ++ni) { v8f z = {}; accG[mi][ni] = z; accU[mi][ni] = z; }

    // ---- prologue: stage tile 0 into buffer 0 ----
    issue_x(0, 0);
    {
        WStage w0 = load_w(0);
        store_w(0, w0);
    }
    WAIT_ASYNC();
    __syncthreads();

    const int nk = HDIM / TK;
    for (int it = 0; it < nk; ++it) {
        const int cur = it & 1, nxt = cur ^ 1;
        const bool pre = (it + 1) < nk;
        const int kbn = (it + 1) * TK;

        WStage wn{};
        if (pre) {
            issue_x(nxt, kbn);          // async: covered by compute below
            wn = load_w(kbn);           // reg-staged fp32 weights
            if (it + 2 < nk)
                __builtin_prefetch((const void*)(Wp + (long)((it + 2) * TK + wk2) * ldw + n0 + wnn), 0, 1);
        }

        // ---- compute current tile (16-bit A/B VGPR layout, ISA 7.12.2) ----
        const int krow  = lane & 15;
        const int khalf = (lane < 16) ? 0 : 8;
        AFrag a[2], bg[2], bu[2];
#pragma unroll
        for (int mi = 0; mi < 2; ++mi) {
            const int row = mw * 32 + mi * 16 + krow;
            a[mi].h[0] = *(const v8bf*)&Xs[cur][row * LDS_LD + khalf];
            a[mi].h[1] = *(const v8bf*)&Xs[cur][row * LDS_LD + khalf + 16];
        }
#pragma unroll
        for (int ni = 0; ni < 2; ++ni) {
            const int col = nw * 32 + ni * 16 + krow;
            bg[ni].h[0] = *(const v8bf*)&Gs[cur][col * LDS_LD + khalf];
            bg[ni].h[1] = *(const v8bf*)&Gs[cur][col * LDS_LD + khalf + 16];
            bu[ni].h[0] = *(const v8bf*)&Us[cur][col * LDS_LD + khalf];
            bu[ni].h[1] = *(const v8bf*)&Us[cur][col * LDS_LD + khalf + 16];
        }
#pragma unroll
        for (int mi = 0; mi < 2; ++mi)
#pragma unroll
            for (int ni = 0; ni < 2; ++ni) {
                accG[mi][ni] = __builtin_amdgcn_wmma_f32_16x16x32_bf16(
                    false, a[mi].v, false, bg[ni].v, (short)0, accG[mi][ni], false, false);
                accU[mi][ni] = __builtin_amdgcn_wmma_f32_16x16x32_bf16(
                    false, a[mi].v, false, bu[ni].v, (short)0, accU[mi][ni], false, false);
            }

        if (pre) store_w(nxt, wn);      // convert + LDS store after compute
        WAIT_ASYNC();
        __syncthreads();
    }

    // ---- epilogue: act = scale * silu(g) * u, bf16 store ----
    const int rlo = (lane < 16) ? 0 : 8;
    const int ncl = lane & 15;
#pragma unroll
    for (int mi = 0; mi < 2; ++mi) {
#pragma unroll
        for (int ni = 0; ni < 2; ++ni) {
            const int mbase = m0 + mw * 32 + mi * 16;
            const int nbase = n0 + nw * 32 + ni * 16;
#pragma unroll
            for (int r = 0; r < 8; ++r) {
                const int t = mbase + rlo + r;
                const float s   = sc[(long)t * scale_stride];
                const float g   = accG[mi][ni][r];
                const float u   = accU[mi][ni][r];
                const float sig = 1.f / (1.f + __expf(-g));
                Aout[(long)t * KK_TOT + outc0 + nbase + ncl] = (bf16)(s * g * sig * u);
            }
        }
    }
}

// ---------------------------------------------------------------------------
// Kernel 4: down GEMM. out[T,H] = A[T,KK] @ concat(w_shared_down, w_exp_down)
// Gate/combine weights already folded into A. Segment boundary is TK-aligned.
// ---------------------------------------------------------------------------
__global__ __launch_bounds__(256)
void down_gemm_kernel(const bf16*  __restrict__ A,     // [T, KK_TOT] bf16
                      const float* __restrict__ Bsha,  // [IS_SZ, H]
                      const float* __restrict__ Bexp,  // [NEXP*IE_SZ, H]
                      float* __restrict__ Out)         // [T, H]
{
    __shared__ bf16 As[2][TM * LDS_LD];
    __shared__ bf16 Bs[2][TN * LDS_LD];

    const int n0 = blockIdx.x * TN;
    const int m0 = blockIdx.y * TM;

    const int tid  = threadIdx.x;
    const int lane = tid & 31;
    const int wv   = tid >> 5;
    const int mw   = wv & 3;
    const int nw   = wv >> 2;

    const int xr = tid >> 2;
    const int xk = (tid & 3) * 8;
    const bf16* ag0 = A + (long)(m0 + xr) * KK_TOT + xk;
    const bf16* ag1 = A + (long)(m0 + xr + 64) * KK_TOT + xk;

    const int wk2 = (tid >> 4) * 2;
    const int wnn = (tid & 15) * 4;

    auto issue_a = [&](int s, int kb) {
        ASYNC_CP16(ag0 + kb, &As[s][xr * LDS_LD + xk]);
        ASYNC_CP16(ag1 + kb, &As[s][(xr + 64) * LDS_LD + xk]);
    };

    struct BStage { float4 b0, b1; };
    auto load_b = [&](int kb) {
        const float* base = (kb < IS_SZ) ? (Bsha + (long)kb * HDIM)
                                         : (Bexp + (long)(kb - IS_SZ) * HDIM);
        const float* r0 = base + (long)wk2 * HDIM + n0 + wnn;
        BStage v;
        v.b0 = *(const float4*)(r0);
        v.b1 = *(const float4*)(r0 + HDIM);
        return v;
    };
    auto store_b = [&](int s, const BStage& v) {
        bf16* bp = &Bs[s][wnn * LDS_LD + wk2];
        PkBf t;
        t.h[0] = (bf16)v.b0.x; t.h[1] = (bf16)v.b1.x; *(unsigned int*)(bp + 0 * LDS_LD) = t.u;
        t.h[0] = (bf16)v.b0.y; t.h[1] = (bf16)v.b1.y; *(unsigned int*)(bp + 1 * LDS_LD) = t.u;
        t.h[0] = (bf16)v.b0.z; t.h[1] = (bf16)v.b1.z; *(unsigned int*)(bp + 2 * LDS_LD) = t.u;
        t.h[0] = (bf16)v.b0.w; t.h[1] = (bf16)v.b1.w; *(unsigned int*)(bp + 3 * LDS_LD) = t.u;
    };

    v8f acc[2][2];
#pragma unroll
    for (int mi = 0; mi < 2; ++mi)
#pragma unroll
        for (int ni = 0; ni < 2; ++ni) { v8f z = {}; acc[mi][ni] = z; }

    issue_a(0, 0);
    {
        BStage b0 = load_b(0);
        store_b(0, b0);
    }
    WAIT_ASYNC();
    __syncthreads();

    const int nk = KK_TOT / TK;
    for (int it = 0; it < nk; ++it) {
        const int cur = it & 1, nxt = cur ^ 1;
        const bool pre = (it + 1) < nk;
        const int kbn = (it + 1) * TK;

        BStage bn{};
        if (pre) {
            issue_a(nxt, kbn);
            bn = load_b(kbn);
        }

        const int krow  = lane & 15;
        const int khalf = (lane < 16) ? 0 : 8;
        AFrag af[2], bf_[2];
#pragma unroll
        for (int mi = 0; mi < 2; ++mi) {
            const int row = mw * 32 + mi * 16 + krow;
            af[mi].h[0] = *(const v8bf*)&As[cur][row * LDS_LD + khalf];
            af[mi].h[1] = *(const v8bf*)&As[cur][row * LDS_LD + khalf + 16];
        }
#pragma unroll
        for (int ni = 0; ni < 2; ++ni) {
            const int col = nw * 32 + ni * 16 + krow;
            bf_[ni].h[0] = *(const v8bf*)&Bs[cur][col * LDS_LD + khalf];
            bf_[ni].h[1] = *(const v8bf*)&Bs[cur][col * LDS_LD + khalf + 16];
        }
#pragma unroll
        for (int mi = 0; mi < 2; ++mi)
#pragma unroll
            for (int ni = 0; ni < 2; ++ni)
                acc[mi][ni] = __builtin_amdgcn_wmma_f32_16x16x32_bf16(
                    false, af[mi].v, false, bf_[ni].v, (short)0, acc[mi][ni], false, false);

        if (pre) store_b(nxt, bn);
        WAIT_ASYNC();
        __syncthreads();
    }

    const int rlo = (lane < 16) ? 0 : 8;
    const int ncl = lane & 15;
#pragma unroll
    for (int mi = 0; mi < 2; ++mi)
#pragma unroll
        for (int ni = 0; ni < 2; ++ni) {
            const int mbase = m0 + mw * 32 + mi * 16;
            const int nbase = n0 + nw * 32 + ni * 16;
#pragma unroll
            for (int r = 0; r < 8; ++r)
                Out[(long)(mbase + rlo + r) * HDIM + nbase + ncl] = acc[mi][ni][r];
        }
}

// ---------------------------------------------------------------------------
// Host launcher
// ---------------------------------------------------------------------------
extern "C" void kernel_launch(void* const* d_in, const int* in_sizes, int n_in,
                              void* d_out, int out_size, void* d_ws, size_t ws_size,
                              hipStream_t stream)
{
    (void)in_sizes; (void)n_in; (void)out_size; (void)ws_size;

    const float* x       = (const float*)d_in[0]; // [T, H]
    const float* w_gate  = (const float*)d_in[1]; // [H, 8]
    const float* w_sgate = (const float*)d_in[2]; // [H, 1]
    const float* w_sgu   = (const float*)d_in[3]; // [H, 2*I_S]
    const float* w_sdown = (const float*)d_in[4]; // [I_S, H]
    const float* w_egu   = (const float*)d_in[5]; // [E, H, 2*I_E]
    const float* w_edown = (const float*)d_in[6]; // [E, I_E, H]
    float* out = (float*)d_out;

    char*  ws       = (char*)d_ws;
    float* comb     = (float*)(ws);                                    // 32 KB
    float* gate_sig = (float*)(ws + 32768);                            //  4 KB
    bf16*  xb       = (bf16*) (ws + 36864);                            //  4 MB
    bf16*  Aact     = (bf16*) (ws + 36864 + (size_t)T_TOK * HDIM * 2); // 34.6 MB

    router_kernel<<<T_TOK / 4, 128, 0, stream>>>(x, w_gate, w_sgate, comb, gate_sig);

    {
        const long n = (long)T_TOK * HDIM;
        cvt_bf16_kernel<<<(unsigned)(n / (256 * 4)), 256, 0, stream>>>(x, xb, n);
    }

    // shared expert SwiGLU -> A[:, 0:I_S), scale = sigmoid gate
    gu_act_kernel<<<dim3(IS_SZ / TN, T_TOK / TM, 1), 256, 0, stream>>>(
        xb, w_sgu, 0L, IS_SZ, gate_sig, 1, Aact, 0, 0);

    // routed experts SwiGLU -> A[:, I_S + e*I_E), scale = comb[t, e]
    gu_act_kernel<<<dim3(IE_SZ / TN, T_TOK / TM, NEXP), 256, 0, stream>>>(
        xb, w_egu, (long)HDIM * 2 * IE_SZ, IE_SZ, comb, NEXP, Aact, IS_SZ, IE_SZ);

    // fused down GEMM -> out
    down_gemm_kernel<<<dim3(HDIM / TN, T_TOK / TM), 256, 0, stream>>>(
        Aact, w_sdown, w_edown, out);
}
